// BasicModel_8847632630249
// MI455X (gfx1250) — matrix operand order repaired and checked by reading, not compiled, not run on hardware
//
#include <hip/hip_runtime.h>

// ---------------------------------------------------------------------------
// MI455X (gfx1250) implementation of the biLSTM seq2seq scorer.
// All matrix math uses v_wmma_f32_16x16x32_bf16 (bf16 in, fp32 accum).
//
// Assumed input flattening order (setup_inputs dict, leaves in order):
//  0 state[64,4,128] i32   1 instruction[64,1,64] i32   2 emb[32000,256] f32
//  3 i_wih_f[2048,256] 4 i_whh_f[2048,512] 5 i_b_f[2048]
//  6 i_wih_b 7 i_whh_b 8 i_b_b
//  9 s_wih_f[4,2048,256] 10 s_whh_f[4,2048,512] 11 s_b_f[4,2048]
// 12 s_wih_b 13 s_whh_b 14 s_b_b
// 15 comb_w[1024,4096] 16 comb_b[1024]
// 17 t_att_w[1,2048] 18 t_att_b[1] 19 t_w[64,2048] 20 t_b[64]
// 21 o1_att_w 22 o1_att_b 23 o1_w[2000,2112] 24 o1_b[2000]
// 25 o2_att_w 26 o2_att_b 27 o2_w[2000,4112] 28 o2_b[2000]
// ---------------------------------------------------------------------------

typedef __attribute__((ext_vector_type(16))) __bf16 v16bf;
typedef __attribute__((ext_vector_type(8)))  __bf16 v8bf;
typedef __attribute__((ext_vector_type(8)))  float  v8f;

#define BATCH 64
#define EMB   256
#define HID   512
#define HSTRIDE 536            // bf16 elems per LDS h row (16B aligned, bank-skewed)
#define XSTRIDE 264            // bf16 elems per LDS x row
#define KB_LSTM 24             // (256 + 512) / 32 k-blocks

__device__ __forceinline__ v16bf mk16(v8bf lo, v8bf hi) {
  v16bf r;
#pragma unroll
  for (int i = 0; i < 8; ++i) { r[i] = lo[i]; r[i + 8] = hi[i]; }
  return r;
}
__device__ __forceinline__ v16bf ldA(const __bf16* p) {       // A fragment
  return mk16(*(const v8bf*)p, *(const v8bf*)(p + 16));
}
__device__ __forceinline__ v16bf ldB(const __bf16* p) {       // packed B fragment
  return mk16(*(const v8bf*)p, *(const v8bf*)(p + 8));
}
__device__ __forceinline__ v8f zero8() {
  v8f z;
#pragma unroll
  for (int i = 0; i < 8; ++i) z[i] = 0.0f;
  return z;
}
__device__ __forceinline__ float sigm(float x) { return 1.0f / (1.0f + expf(-x)); }

// ---------------------------------------------------------------------------
// Weight packers: W[N,K] row-major fp32 -> bf16 tiles pk[gt][kb][lane][16]
// with pk[...][lane][s] = W[gt*16+lane%16][kb*32+16*(s/8)+8*(lane/16)+s%8]
// ---------------------------------------------------------------------------
__global__ __launch_bounds__(256) void pack_lstm_kernel(
    const float* __restrict__ wih, const float* __restrict__ whh,
    __bf16* __restrict__ dst) {
  const int total = 128 * KB_LSTM * 512;          // 2048 x 768
  int idx = blockIdx.x * 256 + threadIdx.x;
  if (idx >= total) return;
  int i    = idx & 15;
  int lane = (idx >> 4) & 31;
  int kb   = (idx >> 9) % KB_LSTM;
  int gt   = idx / (KB_LSTM * 512);
  int n = gt * 16 + (lane & 15);
  int k = kb * 32 + ((i >> 3) << 4) + (lane >> 4) * 8 + (i & 7);
  float v = (k < EMB) ? wih[(size_t)n * EMB + k]
                      : whh[(size_t)n * HID + (k - EMB)];
  dst[idx] = (__bf16)v;
}

__global__ __launch_bounds__(256) void pack_generic_kernel(
    const float* __restrict__ W, __bf16* __restrict__ dst,
    int Nsrc, int Ksrc, int KB, int Nt) {
  long long total = (long long)Nt * KB * 512;
  long long idx = (long long)blockIdx.x * 256 + threadIdx.x;
  if (idx >= total) return;
  int i    = (int)(idx & 15);
  int lane = (int)((idx >> 4) & 31);
  long long rest = idx >> 9;
  int kb = (int)(rest % KB);
  int gt = (int)(rest / KB);
  int n = gt * 16 + (lane & 15);
  int k = kb * 32 + ((i >> 3) << 4) + (lane >> 4) * 8 + (i & 7);
  float v = (n < Nsrc && k < Ksrc) ? W[(size_t)n * Ksrc + k] : 0.0f;
  dst[idx] = (__bf16)v;
}

// ---------------------------------------------------------------------------
// Embedding gathers (fp32 emb -> bf16 time-major activations)
// ---------------------------------------------------------------------------
__global__ __launch_bounds__(256) void gather_instr_kernel(
    const int* __restrict__ ins, const float* __restrict__ emb,
    __bf16* __restrict__ xi) {
  int e = threadIdx.x;
  int row = blockIdx.x;            // t*64 + b
  int t = row >> 6, b = row & 63;
  int tok = ins[b * 64 + t];       // instruction[B,1,T]
  xi[(size_t)row * EMB + e] = (__bf16)emb[(size_t)tok * EMB + e];
}
__global__ __launch_bounds__(256) void gather_state_kernel(
    const int* __restrict__ st, const float* __restrict__ emb,
    __bf16* __restrict__ xs) {
  int e = threadIdx.x;
  int row = blockIdx.x;            // ch*8192 + s*64 + b
  int ch = row >> 13, rem = row & 8191;
  int s = rem >> 6, b = rem & 63;
  int tok = st[(b * 4 + ch) * 128 + s];   // state[B,4,S]
  xs[(size_t)row * EMB + e] = (__bf16)emb[(size_t)tok * EMB + e];
}

// ---------------------------------------------------------------------------
// Persistent LSTM: one 32-wave block per sequence-direction.
// Wave w owns h columns [w*16, w*16+16); per step it computes gate tiles
// (i,f,g,o) = rows of [x_t ; h] (K=768) times packed weights, all via WMMA.
// h (single buffer) and x_t (double-buffered stage) live in LDS bf16;
// c stays in VGPRs.  Row-tiles processed in pairs so each B fragment feeds
// two WMMAs (halves weight re-streaming).  Per step:
//   issue next-x global loads -> compute (pure LDS reads + weight streaming)
//   -> barrier -> commit new h + staged x to LDS -> barrier.
// ---------------------------------------------------------------------------
struct LstmSeq {
  const __bf16* x;      // [L*64, 256]
  const __bf16* wpk;    // packed [128][24][32][16] bf16
  const float*  bias;   // [2048]
  float*        out;    // enc (mode 0) or st_cat (mode 1)
  int steps, L, dir, mode, col0;
};
struct LstmArgs { LstmSeq s[10]; };

__global__ __launch_bounds__(1024, 1) void lstm_kernel(LstmArgs args) {
  __shared__ __bf16 hbuf[BATCH][HSTRIDE];        // 68,608 B
  __shared__ __bf16 xstage[2][BATCH][XSTRIDE];   // 67,584 B  (total ~136 KB)
  const LstmSeq sq = args.s[blockIdx.x];
  const int tid  = threadIdx.x;
  const int w    = tid >> 5;        // wave id = h column tile
  const int lane = tid & 31;
  const int ln   = lane & 15;
  const int hi   = (lane >> 4) & 1;

  // zero h
  for (int i = tid; i < BATCH * HSTRIDE; i += 1024)
    ((__bf16*)hbuf)[i] = (__bf16)0.0f;

  // stage x for the first step (each thread owns 32B of the 32KB tile)
  const int srow = tid >> 4;          // 0..63
  const int scol = (tid & 15) * 16;   // element offset
  {
    const int t0 = sq.dir ? (sq.L - 1) : 0;
    const __bf16* g = sq.x + ((size_t)t0 * BATCH + srow) * EMB + scol;
    v8bf a = *(const v8bf*)g;
    v8bf b = *(const v8bf*)(g + 8);
    __bf16* d = &xstage[0][srow][scol];
    *(v8bf*)d = a;
    *(v8bf*)(d + 8) = b;
  }
  __syncthreads();

  v8f c[4];
#pragma unroll
  for (int r = 0; r < 4; ++r) c[r] = zero8();

  const int col = w * 16 + ln;
  const float bi0 = sq.bias[0 * HID + col];
  const float bf0 = sq.bias[1 * HID + col];
  const float bg0 = sq.bias[2 * HID + col];
  const float bo0 = sq.bias[3 * HID + col];

  // per-wave packed weight base: gate g fragment for k-block kb lives at
  // wbase + g*(32*KB_LSTM*512) + kb*512  (constant offsets -> immediates)
  const __bf16* wbase = sq.wpk + ((size_t)w * KB_LSTM) * 512 + lane * 16;
  const size_t GOFF = (size_t)32 * KB_LSTM * 512;

  int curx = 0;
  for (int s = 0; s < sq.steps; ++s) {
    const int time = sq.dir ? (sq.L - 1 - s) : s;

    // issue next step's x loads early (latency hidden behind WMMA work)
    v8bf sx0, sx1;
    const bool do_stage = (s + 1 < sq.steps);
    if (do_stage) {
      const int tn = sq.dir ? (sq.L - 2 - s) : (s + 1);
      const __bf16* g = sq.x + ((size_t)tn * BATCH + srow) * EMB + scol;
      sx0 = *(const v8bf*)g;
      sx1 = *(const v8bf*)(g + 8);
    }

    const __bf16* xb = &xstage[curx][0][0];
    float hval[4][8];

#pragma unroll
    for (int rp = 0; rp < 2; ++rp) {
      const int r0 = rp * 2;
      v8f acc[2][4];
#pragma unroll
      for (int rr = 0; rr < 2; ++rr)
#pragma unroll
        for (int g = 0; g < 4; ++g) acc[rr][g] = zero8();

      const __bf16* xrow0 = xb + (size_t)(r0 * 16 + ln) * XSTRIDE + hi * 8;
      const __bf16* xrow1 = xrow0 + (size_t)16 * XSTRIDE;
      const __bf16* hrow0 =
          &hbuf[0][0] + (size_t)(r0 * 16 + ln) * HSTRIDE + hi * 8;
      const __bf16* hrow1 = hrow0 + (size_t)16 * HSTRIDE;

      // ---- x phase: K = 0..255 (LDS loads) ----
      for (int kb = 0; kb < 8; ++kb) {
        v16bf a0 = ldA(xrow0 + kb * 32);
        v16bf a1 = ldA(xrow1 + kb * 32);
        const __bf16* bp = wbase + (size_t)kb * 512;
#pragma unroll
        for (int g = 0; g < 4; ++g) {
          v16bf bm = ldB(bp + (size_t)g * GOFF);
          acc[0][g] = __builtin_amdgcn_wmma_f32_16x16x32_bf16(
              false, a0, false, bm, (short)0, acc[0][g], false, false);
          acc[1][g] = __builtin_amdgcn_wmma_f32_16x16x32_bf16(
              false, a1, false, bm, (short)0, acc[1][g], false, false);
        }
      }
      // ---- h phase: K = 256..767 (LDS loads) ----
      for (int kb = 0; kb < 16; ++kb) {
        v16bf a0 = ldA(hrow0 + kb * 32);
        v16bf a1 = ldA(hrow1 + kb * 32);
        const __bf16* bp = wbase + (size_t)(kb + 8) * 512;
#pragma unroll
        for (int g = 0; g < 4; ++g) {
          v16bf bm = ldB(bp + (size_t)g * GOFF);
          acc[0][g] = __builtin_amdgcn_wmma_f32_16x16x32_bf16(
              false, a0, false, bm, (short)0, acc[0][g], false, false);
          acc[1][g] = __builtin_amdgcn_wmma_f32_16x16x32_bf16(
              false, a1, false, bm, (short)0, acc[1][g], false, false);
        }
      }

      // ---- gate nonlinearity + state update (h kept in regs until barrier)
#pragma unroll
      for (int rr = 0; rr < 2; ++rr) {
        const int r = r0 + rr;
#pragma unroll
        for (int v = 0; v < 8; ++v) {
          float iv = sigm(acc[rr][0][v] + bi0);
          float fv = sigm(acc[rr][1][v] + bf0);
          float gv = tanhf(acc[rr][2][v] + bg0);
          float ov = sigm(acc[rr][3][v] + bo0);
          float cn = fv * c[r][v] + iv * gv;
          c[r][v] = cn;
          hval[r][v] = ov * tanhf(cn);
        }
        if (sq.mode == 0) {          // enc[t,B,2H] every step
          float* orow =
              sq.out + ((size_t)time * BATCH + r * 16 + hi * 8) * (2 * HID) +
              sq.col0 + col;
#pragma unroll
          for (int v = 0; v < 8; ++v) orow[(size_t)v * (2 * HID)] = hval[r][v];
        } else if (s == sq.steps - 1) {  // st_cat[B,8H] last step only
          float* orow =
              sq.out + (size_t)(r * 16 + hi * 8) * (8 * HID) + sq.col0 + col;
#pragma unroll
          for (int v = 0; v < 8; ++v) orow[(size_t)v * (8 * HID)] = hval[r][v];
        }
      }
    }

    __syncthreads();   // all LDS reads of h and x done
    // commit new h (D layout: row = r*16 + v + hi*8) and staged x
#pragma unroll
    for (int r = 0; r < 4; ++r)
#pragma unroll
      for (int v = 0; v < 8; ++v)
        hbuf[r * 16 + v + hi * 8][col] = (__bf16)hval[r][v];
    if (do_stage) {
      __bf16* d = &xstage[curx ^ 1][srow][scol];
      *(v8bf*)d = sx0;
      *(v8bf*)(d + 8) = sx1;
    }
    __syncthreads();
    curx ^= 1;
  }
}

// ---------------------------------------------------------------------------
// Generic WMMA GEMM: C[64,N] = A[64,K]bf16 * Bpk + bias.  4 waves = 4 M-tiles,
// one N-tile per block.
// ---------------------------------------------------------------------------
__global__ __launch_bounds__(128) void gemm_kernel(
    const __bf16* __restrict__ A, int lda,
    const __bf16* __restrict__ Bpk, const float* __restrict__ bias,
    float* __restrict__ C, int ldc, int KB, int N) {
  const int wv   = threadIdx.x >> 5;
  const int lane = threadIdx.x & 31;
  const int ln   = lane & 15;
  const int hi   = (lane >> 4) & 1;
  const int nt   = blockIdx.x;
  v8f acc = zero8();
  const __bf16* arow = A + (size_t)(wv * 16 + ln) * lda + hi * 8;
  const __bf16* bbase = Bpk + ((size_t)nt * KB) * 512 + lane * 16;
  for (int kb = 0; kb < KB; ++kb) {
    v16bf a = ldA(arow + kb * 32);
    v16bf b = ldB(bbase + (size_t)kb * 512);
    acc = __builtin_amdgcn_wmma_f32_16x16x32_bf16(
        false, a, false, b, (short)0, acc, false, false);
  }
  const int colo = nt * 16 + ln;
  if (colo < N) {
    const float bv = bias[colo];
#pragma unroll
    for (int v = 0; v < 8; ++v) {
      int m = wv * 16 + v + hi * 8;
      C[(size_t)m * ldc + colo] = acc[v] + bv;
    }
  }
}

// ---------------------------------------------------------------------------
// Attention scorer: scores over T, softmax, attended vector -> bf16 into the
// "full" feature row (cols 0..1023).  One block per batch element.
// ---------------------------------------------------------------------------
__global__ __launch_bounds__(256) void att_kernel(
    const float* __restrict__ enc, const float* __restrict__ stv,
    const float* __restrict__ aw, const float* __restrict__ ab,
    __bf16* __restrict__ full, int ldf) {
  __shared__ float red[256];
  __shared__ float sc[64];
  __shared__ float wt[64];
  const int b = blockIdx.x, tid = threadIdx.x;

  float p = 0.0f;
  for (int e = tid; e < 1024; e += 256) p += stv[b * 1024 + e] * aw[e];
  red[tid] = p; __syncthreads();
  for (int s = 128; s > 0; s >>= 1) { if (tid < s) red[tid] += red[tid + s]; __syncthreads(); }
  const float s1 = red[0] + ab[0];
  __syncthreads();

  for (int t = 0; t < 64; ++t) {
    const float* er = enc + ((size_t)t * 64 + b) * 1024;
    float q = 0.0f;
    for (int e = tid; e < 1024; e += 256) q += er[e] * aw[1024 + e];
    red[tid] = q; __syncthreads();
    for (int s = 128; s > 0; s >>= 1) { if (tid < s) red[tid] += red[tid + s]; __syncthreads(); }
    if (tid == 0) sc[t] = s1 + red[0];
    __syncthreads();
  }
  if (tid == 0) {
    float m = sc[0];
    for (int t = 1; t < 64; ++t) m = fmaxf(m, sc[t]);
    float sm = 0.0f;
    for (int t = 0; t < 64; ++t) { float x = expf(sc[t] - m); wt[t] = x; sm += x; }
    float inv = 1.0f / sm;
    for (int t = 0; t < 64; ++t) wt[t] *= inv;
  }
  __syncthreads();
  for (int e = tid; e < 1024; e += 256) {
    float a = 0.0f;
    for (int t = 0; t < 64; ++t) a += wt[t] * enc[((size_t)t * 64 + b) * 1024 + e];
    full[(size_t)b * ldf + e] = (__bf16)a;
  }
}

// ---------------------------------------------------------------------------
// Small glue kernels
// ---------------------------------------------------------------------------
__global__ __launch_bounds__(256) void f32_to_bf16_kernel(
    const float* __restrict__ src, __bf16* __restrict__ dst, int n) {
  int i = blockIdx.x * 256 + threadIdx.x;
  if (i < n) dst[i] = (__bf16)src[i];
}
__global__ __launch_bounds__(256) void st_broadcast_kernel(
    const float* __restrict__ es, __bf16* __restrict__ fT,
    __bf16* __restrict__ fO1, __bf16* __restrict__ fO2) {
  int c = (blockIdx.x & 3) * 256 + threadIdx.x;
  int b = blockIdx.x >> 2;
  __bf16 h = (__bf16)es[b * 1024 + c];
  fT [(size_t)b * 2048 + 1024 + c] = h;
  fO1[(size_t)b * 2112 + 1024 + c] = h;
  fO2[(size_t)b * 4128 + 1024 + c] = h;
}
__global__ __launch_bounds__(256) void qt_broadcast_kernel(
    const float* __restrict__ qt, __bf16* __restrict__ fO1,
    __bf16* __restrict__ fO2) {
  int i = blockIdx.x * 256 + threadIdx.x;
  if (i >= 64 * 64) return;
  int b = i >> 6, c = i & 63;
  __bf16 h = (__bf16)qt[i];
  fO1[(size_t)b * 2112 + 2048 + c] = h;
  fO2[(size_t)b * 4128 + 2048 + c] = h;
}
__global__ __launch_bounds__(256) void qo1_broadcast_kernel(
    const float* __restrict__ q, __bf16* __restrict__ fO2) {
  int i = blockIdx.x * 256 + threadIdx.x;
  if (i >= 64 * 2016) return;
  int b = i / 2016, c = i % 2016;
  float v = (c < 2000) ? q[b * 2000 + c] : 0.0f;   // zero-pads K to 4128
  fO2[(size_t)b * 4128 + 2112 + c] = (__bf16)v;
}
__global__ __launch_bounds__(256) void logsoftmax_kernel(
    const float* __restrict__ q, float* __restrict__ out, int C) {
  __shared__ float red[256];
  const int b = blockIdx.x, tid = threadIdx.x;
  const float* row = q + (size_t)b * C;
  float m = -1e30f;
  for (int c = tid; c < C; c += 256) m = fmaxf(m, row[c]);
  red[tid] = m; __syncthreads();
  for (int s = 128; s > 0; s >>= 1) { if (tid < s) red[tid] = fmaxf(red[tid], red[tid + s]); __syncthreads(); }
  m = red[0]; __syncthreads();
  float p = 0.0f;
  for (int c = tid; c < C; c += 256) p += expf(row[c] - m);
  red[tid] = p; __syncthreads();
  for (int s = 128; s > 0; s >>= 1) { if (tid < s) red[tid] += red[tid + s]; __syncthreads(); }
  const float lse = m + logf(red[0]);
  float* orow = out + (size_t)b * C;
  for (int c = tid; c < C; c += 256) orow[c] = row[c] - lse;
}

// ---------------------------------------------------------------------------
extern "C" void kernel_launch(void* const* d_in, const int* in_sizes, int n_in,
                              void* d_out, int out_size, void* d_ws, size_t ws_size,
                              hipStream_t stream) {
  (void)in_sizes; (void)n_in; (void)out_size; (void)ws_size;
  const int*   state    = (const int*)d_in[0];
  const int*   instr    = (const int*)d_in[1];
  const float* emb      = (const float*)d_in[2];
  const float* i_wih_f  = (const float*)d_in[3];
  const float* i_whh_f  = (const float*)d_in[4];
  const float* i_b_f    = (const float*)d_in[5];
  const float* i_wih_b  = (const float*)d_in[6];
  const float* i_whh_b  = (const float*)d_in[7];
  const float* i_b_b    = (const float*)d_in[8];
  const float* s_wih_f  = (const float*)d_in[9];
  const float* s_whh_f  = (const float*)d_in[10];
  const float* s_b_f    = (const float*)d_in[11];
  const float* s_wih_b  = (const float*)d_in[12];
  const float* s_whh_b  = (const float*)d_in[13];
  const float* s_b_b    = (const float*)d_in[14];
  const float* comb_w   = (const float*)d_in[15];
  const float* comb_b   = (const float*)d_in[16];
  const float* t_att_w  = (const float*)d_in[17];
  const float* t_att_b  = (const float*)d_in[18];
  const float* t_w      = (const float*)d_in[19];
  const float* t_b      = (const float*)d_in[20];
  const float* o1_att_w = (const float*)d_in[21];
  const float* o1_att_b = (const float*)d_in[22];
  const float* o1_w     = (const float*)d_in[23];
  const float* o1_b     = (const float*)d_in[24];
  const float* o2_att_w = (const float*)d_in[25];
  const float* o2_att_b = (const float*)d_in[26];
  const float* o2_w     = (const float*)d_in[27];
  const float* o2_b     = (const float*)d_in[28];

  // ---- workspace carving (~104 MB) ----
  char* ws = (char*)d_ws;
  size_t off = 0;
  auto carve = [&](size_t bytes) -> char* {
    char* p = ws + off;
    off += (bytes + 255) & ~(size_t)255;
    return p;
  };
  const size_t LSTM_PK = 2048ull * 768;                     // elems / sequence
  __bf16* wpk_lstm = (__bf16*)carve(10ull * LSTM_PK * 2);
  __bf16* wpk_comb = (__bf16*)carve(64ull * 128 * 512 * 2); // N=1024 K=4096
  __bf16* wpk_t    = (__bf16*)carve(4ull * 64 * 512 * 2);   // N=64   K=2048
  __bf16* wpk_o1   = (__bf16*)carve(126ull * 66 * 512 * 2); // N=2016 K=2112
  __bf16* wpk_o2   = (__bf16*)carve(126ull * 129 * 512 * 2);// N=2016 K=4128
  __bf16* xi       = (__bf16*)carve(4096ull * 256 * 2);
  __bf16* xs       = (__bf16*)carve(32768ull * 256 * 2);
  float*  enc      = (float*)carve(4096ull * 1024 * 4);     // [T*B, 2H]
  float*  stcat    = (float*)carve(64ull * 4096 * 4);       // [B, 8H]
  __bf16* stcat_bf = (__bf16*)carve(64ull * 4096 * 2);
  float*  est      = (float*)carve(64ull * 1024 * 4);       // encoded_state
  __bf16* fullT    = (__bf16*)carve(64ull * 2048 * 2);
  __bf16* fullO1   = (__bf16*)carve(64ull * 2112 * 2);
  __bf16* fullO2   = (__bf16*)carve(64ull * 4128 * 2);
  float*  qt       = (float*)carve(64ull * 64 * 4);
  float*  qo1      = (float*)carve(64ull * 2000 * 4);
  float*  qo2      = (float*)carve(64ull * 2000 * 4);

  // ---- 1. pack weights (fp32 -> bf16 WMMA tiles) ----
  const float* wih_src[10] = {
      i_wih_f, i_wih_b,
      s_wih_f + 0 * 524288, s_wih_f + 1 * 524288, s_wih_f + 2 * 524288, s_wih_f + 3 * 524288,
      s_wih_b + 0 * 524288, s_wih_b + 1 * 524288, s_wih_b + 2 * 524288, s_wih_b + 3 * 524288 };
  const float* whh_src[10] = {
      i_whh_f, i_whh_b,
      s_whh_f + 0 * 1048576, s_whh_f + 1 * 1048576, s_whh_f + 2 * 1048576, s_whh_f + 3 * 1048576,
      s_whh_b + 0 * 1048576, s_whh_b + 1 * 1048576, s_whh_b + 2 * 1048576, s_whh_b + 3 * 1048576 };
  for (int q = 0; q < 10; ++q)
    pack_lstm_kernel<<<6144, 256, 0, stream>>>(wih_src[q], whh_src[q],
                                               wpk_lstm + (size_t)q * LSTM_PK);
  pack_generic_kernel<<<16384, 256, 0, stream>>>(comb_w, wpk_comb, 1024, 4096, 128, 64);
  pack_generic_kernel<<<512,   256, 0, stream>>>(t_w,    wpk_t,    64,   2048, 64,  4);
  pack_generic_kernel<<<16632, 256, 0, stream>>>(o1_w,   wpk_o1,   2000, 2112, 66,  126);
  pack_generic_kernel<<<32508, 256, 0, stream>>>(o2_w,   wpk_o2,   2000, 4112, 129, 126);

  // ---- 2. embedding gathers ----
  gather_instr_kernel<<<4096, 256, 0, stream>>>(instr, emb, xi);
  gather_state_kernel<<<32768, 256, 0, stream>>>(state, emb, xs);

  // ---- 3. all 10 LSTM scans in one persistent kernel ----
  // NOTE: state hs[-1] = concat(hf[S-1], hb[0]) -> backward state LSTMs need 1 step.
  LstmArgs la;
  la.s[0] = { xi, wpk_lstm + 0 * LSTM_PK, i_b_f, enc, 64, 64, 0, 0, 0 };
  la.s[1] = { xi, wpk_lstm + 1 * LSTM_PK, i_b_b, enc, 64, 64, 1, 0, 512 };
  for (int ch = 0; ch < 4; ++ch) {
    la.s[2 + ch] = { xs + (size_t)ch * 8192 * 256, wpk_lstm + (size_t)(2 + ch) * LSTM_PK,
                     s_b_f + ch * 2048, stcat, 128, 128, 0, 1, ch * 1024 };
    la.s[6 + ch] = { xs + (size_t)ch * 8192 * 256, wpk_lstm + (size_t)(6 + ch) * LSTM_PK,
                     s_b_b + ch * 2048, stcat, 1,   128, 1, 1, ch * 1024 + 512 };
  }
  lstm_kernel<<<10, 1024, 0, stream>>>(la);

  // ---- 4. combiner: encoded_state = st_cat @ comb_w.T + b ----
  f32_to_bf16_kernel<<<1024, 256, 0, stream>>>(stcat, stcat_bf, 64 * 4096);
  gemm_kernel<<<64, 128, 0, stream>>>(stcat_bf, 4096, wpk_comb, comb_b, est, 1024, 128, 1024);
  st_broadcast_kernel<<<256, 256, 0, stream>>>(est, fullT, fullO1, fullO2);

  // ---- 5. scorer chain t -> o1 -> o2 ----
  att_kernel<<<64, 256, 0, stream>>>(enc, est, t_att_w, t_att_b, fullT, 2048);
  gemm_kernel<<<4, 128, 0, stream>>>(fullT, 2048, wpk_t, t_b, qt, 64, 64, 64);
  qt_broadcast_kernel<<<16, 256, 0, stream>>>(qt, fullO1, fullO2);

  att_kernel<<<64, 256, 0, stream>>>(enc, est, o1_att_w, o1_att_b, fullO1, 2112);
  gemm_kernel<<<126, 128, 0, stream>>>(fullO1, 2112, wpk_o1, o1_b, qo1, 2000, 66, 2000);
  qo1_broadcast_kernel<<<504, 256, 0, stream>>>(qo1, fullO2);

  att_kernel<<<64, 256, 0, stream>>>(enc, est, o2_att_w, o2_att_b, fullO2, 4128);
  gemm_kernel<<<126, 128, 0, stream>>>(fullO2, 4128, wpk_o2, o2_b, qo2, 2000, 129, 2000);

  // ---- 6. log-softmax into d_out (fp32, concatenated) ----
  float* out = (float*)d_out;
  logsoftmax_kernel<<<64, 256, 0, stream>>>(qt,  out,          64);
  logsoftmax_kernel<<<64, 256, 0, stream>>>(qo1, out + 4096,   2000);
  logsoftmax_kernel<<<64, 256, 0, stream>>>(qo2, out + 132096, 2000);
}
